// GFlowNet_34703335752331
// MI455X (gfx1250) — matrix operand order repaired and checked by reading, not compiled, not run on hardware
//
#include <hip/hip_runtime.h>

// ---------------------------------------------------------------------------
// CDNA5 (gfx1250) implementation of the GFlowNet scan.
// GEMMs use v_wmma_f32_16x16x32_f16 (f16 inputs, f32 accumulate) with
// double-buffered async global->LDS tile staging (ASYNCcnt) when available.
// ---------------------------------------------------------------------------

typedef __attribute__((ext_vector_type(16))) _Float16     v16h;
typedef __attribute__((ext_vector_type(8)))  float        v8f;
typedef __attribute__((ext_vector_type(8)))  unsigned int v8u;

#define B_DIM  512
#define V_DIM  128
#define H_DIM  512
#define LN_EPS 1e-5f

// Padded LDS row stride (halves): 40 halves = 80 B = 20 banks -> the 16
// fragment rows land on 16 distinct bank groups (gcd(20,64)=4, cycle 16).
#define LDS_STRIDE_H 40

// ---- async global->LDS support (probe via __has_builtin; fallback = sync) --
#if defined(__has_builtin)
#  if __has_builtin(__builtin_amdgcn_global_load_async_to_lds_b128) && \
      __has_builtin(__builtin_amdgcn_s_wait_asynccnt)
#    define USE_ASYNC_LDS 1
#  endif
#endif
#ifndef USE_ASYNC_LDS
#  define USE_ASYNC_LDS 0
#endif

// Parameter types per hipcc diagnostic: pointer to 16B int vector in
// global (AS1) / shared (AS3) address spaces.
typedef int vec4i __attribute__((vector_size(4 * sizeof(int))));
typedef __attribute__((address_space(1))) vec4i gas_vec4i;
typedef __attribute__((address_space(3))) vec4i las_vec4i;

__device__ __forceinline__ void copy16_to_lds(const _Float16* g, _Float16* l) {
#if USE_ASYNC_LDS
  __builtin_amdgcn_global_load_async_to_lds_b128(
      (gas_vec4i*)g, (las_vec4i*)l, 0, 0);
#else
  *reinterpret_cast<uint4*>(l) = *reinterpret_cast<const uint4*>(g);
#endif
}

__device__ __forceinline__ void wait_async_lds() {
#if USE_ASYNC_LDS
  __builtin_amdgcn_s_wait_asynccnt(0);
#endif
}

// ---------------------------------------------------------------------------
// Fragment packing helpers (CDNA5 ISA 7.12.2 VGPR layouts)
// ---------------------------------------------------------------------------
__device__ __forceinline__ v16h pack_frag(uint4 a, uint4 b) {
  v8u u;
  u[0] = a.x; u[1] = a.y; u[2] = a.z; u[3] = a.w;
  u[4] = b.x; u[5] = b.y; u[6] = b.z; u[7] = b.w;
  return __builtin_bit_cast(v16h, u);
}

// A-matrix 16x32 f16: lanes 0-15 (M=lane) hold K 0..7 then 16..23;
// lanes 16-31 hold K 8..15 then 24..31.  kh = 0 or 8 halves.
__device__ __forceinline__ v16h load_fragA(const _Float16* rowPtr, int kh) {
  const uint4* p0 = reinterpret_cast<const uint4*>(rowPtr + kh);
  const uint4* p1 = reinterpret_cast<const uint4*>(rowPtr + kh + 16);
  return pack_frag(*p0, *p1);
}

// B-matrix 32x16 f16: lanes 0-15 hold K 0..7 then 16..23;
// lanes 16-31 hold K 8..15 then 24..31.  p already offset by khb (0/8 halves).
__device__ __forceinline__ v16h load_fragB(const _Float16* p) {
  const uint4* q = reinterpret_cast<const uint4*>(p);
  return pack_frag(q[0], q[2]);
}

// ---------------------------------------------------------------------------
// GEMM: C[512,512] = A[512,K] x Bt[512,K]^T + bias  (f16 in, f32 out)
// block = 256 threads (8 waves); wave -> 32x32 patch; WG -> 128x64 tile.
// K-slabs of 32 staged in double-buffered LDS via async loads.
// grid = dim3(512/64, 512/128) = (8,4)
// ---------------------------------------------------------------------------
__global__ __launch_bounds__(256) void gemm_wmma_f16(
    const _Float16* __restrict__ A,   // [512, K] row-major activations
    const _Float16* __restrict__ Bt,  // [512, K] row-major (transposed weights)
    const float* __restrict__ bias,   // [512]
    float* __restrict__ C,            // [512, 512]
    int K)
{
  __shared__ _Float16 lA[2][128 * LDS_STRIDE_H];  // 2 x 10240 B
  __shared__ _Float16 lB[2][64 * LDS_STRIDE_H];   // 2 x  5120 B

  const int t    = threadIdx.x;
  const int lane = t & 31;
  const int wave = t >> 5;
  const int blockM = blockIdx.y * 128;
  const int blockN = blockIdx.x * 64;
  const int l15 = lane & 15;
  const int hi  = lane >> 4;      // 0 or 1
  const int kha = hi << 3;        // A half-offset in halves (0/8)
  const int khb = hi << 3;        // B half-offset in halves (0/8)

  // Cooperative stage-in: A tile = 128 rows x 32 halves = 512 b128 chunks
  // (2 per thread); B tile = 64 rows x 32 halves = 256 chunks (1 per thread).
  const int ca0 = t * 2;
  const int ar0 = ca0 >> 2, ac0 = ca0 & 3;
  const int ca1 = ca0 + 1;
  const int ar1 = ca1 >> 2, ac1 = ca1 & 3;
  const int br  = t >> 2,   bc  = t & 3;

  const int nslab = K >> 5;

  auto stage = [&](int buf, int k) {
    copy16_to_lds(A  + (size_t)(blockM + ar0) * K + k + ac0 * 8,
                  &lA[buf][ar0 * LDS_STRIDE_H + ac0 * 8]);
    copy16_to_lds(A  + (size_t)(blockM + ar1) * K + k + ac1 * 8,
                  &lA[buf][ar1 * LDS_STRIDE_H + ac1 * 8]);
    copy16_to_lds(Bt + (size_t)(blockN + br)  * K + k + bc  * 8,
                  &lB[buf][br  * LDS_STRIDE_H + bc  * 8]);
  };

  stage(0, 0);

  const int waveM = (wave >> 1) * 32;   // 0..96 within tile
  const int waveN = (wave & 1)  * 32;   // 0..32 within tile

  v8f acc00 = {}, acc01 = {}, acc10 = {}, acc11 = {};

  for (int s = 0; s < nslab; ++s) {
    const int buf = s & 1;
    wait_async_lds();       // my slab-s loads have landed in LDS
    __syncthreads();        // everyone's landed; previous buffer reads retired
    if (s + 1 < nslab) stage(buf ^ 1, (s + 1) << 5);

    const _Float16* pa0 = &lA[buf][(waveM + l15)      * LDS_STRIDE_H];
    const _Float16* pa1 = &lA[buf][(waveM + 16 + l15) * LDS_STRIDE_H];
    const _Float16* pb0 = &lB[buf][(waveN + l15)      * LDS_STRIDE_H];
    const _Float16* pb1 = &lB[buf][(waveN + 16 + l15) * LDS_STRIDE_H];

    v16h a0 = load_fragA(pa0, kha);
    v16h a1 = load_fragA(pa1, kha);
    v16h b0 = load_fragB(pb0 + khb);
    v16h b1 = load_fragB(pb1 + khb);

    acc00 = __builtin_amdgcn_wmma_f32_16x16x32_f16(false, a0, false, b0, (short)0, acc00, false, false);
    acc01 = __builtin_amdgcn_wmma_f32_16x16x32_f16(false, a0, false, b1, (short)0, acc01, false, false);
    acc10 = __builtin_amdgcn_wmma_f32_16x16x32_f16(false, a1, false, b0, (short)0, acc10, false, false);
    acc11 = __builtin_amdgcn_wmma_f32_16x16x32_f16(false, a1, false, b1, (short)0, acc11, false, false);
  }

  // C/D layout: VGPR r -> M = r + 8*(lane>=16); N = lane&15.
  const int col0 = blockN + waveN + l15;
  const int col1 = blockN + waveN + 16 + l15;
  const float bv0 = bias[col0];
  const float bv1 = bias[col1];
  const int r0 = blockM + waveM + (hi << 3);
  const int r1 = blockM + waveM + 16 + (hi << 3);
#pragma unroll
  for (int r = 0; r < 8; ++r) {
    C[(size_t)(r0 + r) * H_DIM + col0] = acc00[r] + bv0;
    C[(size_t)(r0 + r) * H_DIM + col1] = acc01[r] + bv1;
    C[(size_t)(r1 + r) * H_DIM + col0] = acc10[r] + bv0;
    C[(size_t)(r1 + r) * H_DIM + col1] = acc11[r] + bv1;
  }
}

// ---------------------------------------------------------------------------
// LayerNorm + ReLU + cast to f16.  One block (256 thr) per row of 512.
// ---------------------------------------------------------------------------
__global__ __launch_bounds__(256) void ln_relu_f16(
    const float* __restrict__ X, const float* __restrict__ g,
    const float* __restrict__ be, _Float16* __restrict__ Y)
{
  __shared__ float s1[256];
  __shared__ float s2[256];
  const int row = blockIdx.x;
  const int t   = threadIdx.x;
  const float x0 = X[(size_t)row * H_DIM + t];
  const float x1 = X[(size_t)row * H_DIM + t + 256];
  s1[t] = x0 + x1;
  s2[t] = x0 * x0 + x1 * x1;
  __syncthreads();
  for (int o = 128; o > 0; o >>= 1) {
    if (t < o) { s1[t] += s1[t + o]; s2[t] += s2[t + o]; }
    __syncthreads();
  }
  const float m   = s1[0] * (1.f / (float)H_DIM);
  const float var = s2[0] * (1.f / (float)H_DIM) - m * m;
  const float inv = rsqrtf(var + LN_EPS);
  float y0 = fmaxf((x0 - m) * inv * g[t]       + be[t],       0.f);
  float y1 = fmaxf((x1 - m) * inv * g[t + 256] + be[t + 256], 0.f);
  Y[(size_t)row * H_DIM + t]       = (_Float16)y0;
  Y[(size_t)row * H_DIM + t + 256] = (_Float16)y1;
}

// ---------------------------------------------------------------------------
// Weight convert + transpose: W[K,N] f32 -> Wt[N,K] f16
// ---------------------------------------------------------------------------
__global__ void transpose_cast(const float* __restrict__ W,
                               _Float16* __restrict__ Wt, int K, int N)
{
  int idx = blockIdx.x * blockDim.x + threadIdx.x;
  if (idx < K * N) {
    int k = idx / N, n = idx % N;
    Wt[(size_t)n * K + k] = (_Float16)W[(size_t)k * N + n];
  }
}

__global__ void zero_f32(float* p, int n) {
  int i = blockIdx.x * blockDim.x + threadIdx.x;
  if (i < n) p[i] = 0.f;
}
__global__ void zero_f16(_Float16* p, int n) {
  int i = blockIdx.x * blockDim.x + threadIdx.x;
  if (i < n) p[i] = (_Float16)0.f;
}

// ---------------------------------------------------------------------------
// qtrunk input: Aq = f16(V * pa_mask[top_order[step]])   (65536 threads)
// ---------------------------------------------------------------------------
__global__ __launch_bounds__(256) void mask_cast(
    const float* __restrict__ V, const int* __restrict__ pa_mask,
    const int* __restrict__ top_order, int step, _Float16* __restrict__ Aq)
{
  int idx = blockIdx.x * 256 + threadIdx.x;        // [0, 512*128)
  int node = top_order[step];
  int j = idx & (V_DIM - 1);
  Aq[idx] = (_Float16)(V[idx] * (float)pa_mask[node * V_DIM + j]);
}

// ---------------------------------------------------------------------------
// qtrunk head + Bernoulli sample + scatter + f16 copy of V_new for fnet.
// One thread per batch element (grid 2 x 256).
// ---------------------------------------------------------------------------
__global__ __launch_bounds__(256) void q_head_sample(
    const _Float16* __restrict__ H3, const float* __restrict__ qW,
    const float* __restrict__ qbias, const float* __restrict__ qmarg,
    const int* __restrict__ top_order, int step,
    const float* __restrict__ u,      // [V_DIM, B_DIM], scanned by step
    const int* __restrict__ pa_mask,
    float* __restrict__ V, _Float16* __restrict__ Af,
    float* __restrict__ lp_buf)
{
  const int b = blockIdx.x * 256 + threadIdx.x;    // 0..511
  const int node = top_order[step];

  float acc = 0.f;
  const _Float16* h = H3 + (size_t)b * H_DIM;
  const float*    w = qW + (size_t)node * H_DIM;
#pragma unroll 8
  for (int k = 0; k < H_DIM; ++k) acc += (float)h[k] * w[k];
  const float x = acc + qbias[node];

  const int* m = pa_mask + node * V_DIM;
  float* vrow = V + (size_t)b * V_DIM;
  float asum = 0.f;
#pragma unroll 8
  for (int j = 0; j < V_DIM; ++j) asum += fabsf(vrow[j]) * (float)m[j];

  const float z = (asum == 0.f) ? qmarg[node] : x;
  const float p = 1.f / (1.f + expf(-z));
  const bool samp = u[step * B_DIM + b] < p;   // u scanned un-permuted

  vrow[node] = samp ? 1.f : -1.f;
  lp_buf[(size_t)step * B_DIM + b] = samp ? logf(p) : log1pf(-p);

  _Float16* a = Af + (size_t)b * V_DIM;
#pragma unroll 8
  for (int j = 0; j < V_DIM; ++j) a[j] = (_Float16)vrow[j];
}

// ---------------------------------------------------------------------------
// fnet head: out[b] = dot(H3f[b,:], f_w4) + f_b4
// ---------------------------------------------------------------------------
__global__ __launch_bounds__(256) void f_head(
    const _Float16* __restrict__ H3, const float* __restrict__ fw4,
    const float* __restrict__ fb4, float* __restrict__ out)
{
  const int b = blockIdx.x * 256 + threadIdx.x;
  float acc = 0.f;
  const _Float16* h = H3 + (size_t)b * H_DIM;
#pragma unroll 8
  for (int k = 0; k < H_DIM; ++k) acc += (float)h[k] * fw4[k];
  out[b] = acc + fb4[0];
}

// ---------------------------------------------------------------------------
// score[b] = V[b]·J·V[b] ; also emits V into d_out (first output).
// ---------------------------------------------------------------------------
__global__ __launch_bounds__(256) void score_kernel(
    const float* __restrict__ V, const float* __restrict__ J,
    float* __restrict__ score, float* __restrict__ outV)
{
  const int b = blockIdx.x * 256 + threadIdx.x;
  const float* v = V + (size_t)b * V_DIM;
  float s = 0.f;
  for (int i = 0; i < V_DIM; ++i) {
    const float vi = v[i];
    float t = 0.f;
#pragma unroll 8
    for (int j = 0; j < V_DIM; ++j) t += J[i * V_DIM + j] * v[j];
    s += vi * t;
  }
  score[b] = s;
#pragma unroll 8
  for (int j = 0; j < V_DIM; ++j) outV[(size_t)b * V_DIM + j] = v[j];
}

// ---------------------------------------------------------------------------
// Detailed-balance loss reduction.
// log_flow = [F0, logFs[0..126], score]; loss = mean((lf[t]+lp[t]-lf[t+1])^2)
// ---------------------------------------------------------------------------
__global__ __launch_bounds__(256) void loss_kernel(
    const float* __restrict__ F0, const float* __restrict__ logFs,
    const float* __restrict__ score, const float* __restrict__ lp,
    float* __restrict__ out)
{
  __shared__ float sm[256];
  const int t = threadIdx.x;
  float local = 0.f;
  for (int b = t; b < B_DIM; b += 256) {
    float prev = F0[b];
    for (int st = 0; st < V_DIM; ++st) {
      float next = (st == V_DIM - 1) ? score[b] : logFs[(size_t)st * B_DIM + b];
      float d = prev + lp[(size_t)st * B_DIM + b] - next;
      local += d * d;
      prev = next;
    }
  }
  sm[t] = local;
  __syncthreads();
  for (int o = 128; o > 0; o >>= 1) {
    if (t < o) sm[t] += sm[t + o];
    __syncthreads();
  }
  if (t == 0) out[0] = sm[0] * (1.f / (float)(V_DIM * B_DIM));
}

// ---------------------------------------------------------------------------
// Host orchestration
// ---------------------------------------------------------------------------
extern "C" void kernel_launch(void* const* d_in, const int* in_sizes, int n_in,
                              void* d_out, int out_size, void* d_ws, size_t ws_size,
                              hipStream_t stream) {
  const float* u      = (const float*)d_in[0];   // [128,512]
  const float* q_w1   = (const float*)d_in[1];
  const float* q_b1   = (const float*)d_in[2];
  const float* q_g1   = (const float*)d_in[3];
  const float* q_be1  = (const float*)d_in[4];
  const float* q_w2   = (const float*)d_in[5];
  const float* q_b2   = (const float*)d_in[6];
  const float* q_g2   = (const float*)d_in[7];
  const float* q_be2  = (const float*)d_in[8];
  const float* q_w3   = (const float*)d_in[9];
  const float* q_b3   = (const float*)d_in[10];
  const float* q_g3   = (const float*)d_in[11];
  const float* q_be3  = (const float*)d_in[12];
  const float* q_W    = (const float*)d_in[13];  // [128,512]
  const float* q_bias = (const float*)d_in[14];
  const float* q_marg = (const float*)d_in[15];
  const float* f_w1   = (const float*)d_in[16];
  const float* f_b1   = (const float*)d_in[17];
  const float* f_g1   = (const float*)d_in[18];
  const float* f_be1  = (const float*)d_in[19];
  const float* f_w2   = (const float*)d_in[20];
  const float* f_b2   = (const float*)d_in[21];
  const float* f_g2   = (const float*)d_in[22];
  const float* f_be2  = (const float*)d_in[23];
  const float* f_w3   = (const float*)d_in[24];
  const float* f_b3   = (const float*)d_in[25];
  const float* f_g3   = (const float*)d_in[26];
  const float* f_be3  = (const float*)d_in[27];
  const float* f_w4   = (const float*)d_in[28];  // [512]
  const float* f_b4   = (const float*)d_in[29];  // [1]
  const float* J      = (const float*)d_in[30];  // [128,128]
  const int* top_order= (const int*)d_in[31];    // [128]
  const int* pa_mask  = (const int*)d_in[32];    // [128,128]
  (void)in_sizes; (void)n_in; (void)out_size; (void)ws_size;

  // ---- workspace carve-up (all 256B aligned) ----
  char* ws = (char*)d_ws;
  size_t off = 0;
  auto carve = [&](size_t bytes) {
    void* p = ws + off;
    off += (bytes + 255) & ~(size_t)255;
    return p;
  };
  _Float16* WQ1T = (_Float16*)carve(H_DIM * V_DIM * 2);   // [512,128]
  _Float16* WQ2T = (_Float16*)carve(H_DIM * H_DIM * 2);
  _Float16* WQ3T = (_Float16*)carve(H_DIM * H_DIM * 2);
  _Float16* WF1T = (_Float16*)carve(H_DIM * V_DIM * 2);
  _Float16* WF2T = (_Float16*)carve(H_DIM * H_DIM * 2);
  _Float16* WF3T = (_Float16*)carve(H_DIM * H_DIM * 2);
  _Float16* AQ   = (_Float16*)carve(B_DIM * V_DIM * 2);   // qtrunk input
  _Float16* AF   = (_Float16*)carve(B_DIM * V_DIM * 2);   // fnet input
  float*    Cbuf = (float*)   carve(B_DIM * H_DIM * 4);
  _Float16* Hbuf = (_Float16*)carve(B_DIM * H_DIM * 2);
  float*    Vbuf = (float*)   carve(B_DIM * V_DIM * 4);
  float*    LP   = (float*)   carve(V_DIM * B_DIM * 4);
  float*    LOGF = (float*)   carve(V_DIM * B_DIM * 4);
  float*    F0b  = (float*)   carve(B_DIM * 4);
  float*    SCb  = (float*)   carve(B_DIM * 4);

  const dim3 blk(256);
  const dim3 ggrid(H_DIM / 64, B_DIM / 128);   // (8,4) for 512x512 C

  // ---- one-time (per call) weight convert/transpose + state init ----
  transpose_cast<<<(V_DIM * H_DIM + 255) / 256, blk, 0, stream>>>(q_w1, WQ1T, V_DIM, H_DIM);
  transpose_cast<<<(H_DIM * H_DIM + 255) / 256, blk, 0, stream>>>(q_w2, WQ2T, H_DIM, H_DIM);
  transpose_cast<<<(H_DIM * H_DIM + 255) / 256, blk, 0, stream>>>(q_w3, WQ3T, H_DIM, H_DIM);
  transpose_cast<<<(V_DIM * H_DIM + 255) / 256, blk, 0, stream>>>(f_w1, WF1T, V_DIM, H_DIM);
  transpose_cast<<<(H_DIM * H_DIM + 255) / 256, blk, 0, stream>>>(f_w2, WF2T, H_DIM, H_DIM);
  transpose_cast<<<(H_DIM * H_DIM + 255) / 256, blk, 0, stream>>>(f_w3, WF3T, H_DIM, H_DIM);
  zero_f32<<<(B_DIM * V_DIM + 255) / 256, blk, 0, stream>>>(Vbuf, B_DIM * V_DIM);
  zero_f16<<<(B_DIM * V_DIM + 255) / 256, blk, 0, stream>>>(AF, B_DIM * V_DIM);

  // ---- F0 = fnet(zeros) ----
  gemm_wmma_f16<<<ggrid, blk, 0, stream>>>(AF, WF1T, f_b1, Cbuf, V_DIM);
  ln_relu_f16<<<B_DIM, blk, 0, stream>>>(Cbuf, f_g1, f_be1, Hbuf);
  gemm_wmma_f16<<<ggrid, blk, 0, stream>>>(Hbuf, WF2T, f_b2, Cbuf, H_DIM);
  ln_relu_f16<<<B_DIM, blk, 0, stream>>>(Cbuf, f_g2, f_be2, Hbuf);
  gemm_wmma_f16<<<ggrid, blk, 0, stream>>>(Hbuf, WF3T, f_b3, Cbuf, H_DIM);
  ln_relu_f16<<<B_DIM, blk, 0, stream>>>(Cbuf, f_g3, f_be3, Hbuf);
  f_head<<<2, blk, 0, stream>>>(Hbuf, f_w4, f_b4, F0b);

  // ---- sequential scan over 128 nodes ----
  for (int step = 0; step < V_DIM; ++step) {
    mask_cast<<<(B_DIM * V_DIM) / 256, blk, 0, stream>>>(Vbuf, pa_mask, top_order, step, AQ);

    gemm_wmma_f16<<<ggrid, blk, 0, stream>>>(AQ, WQ1T, q_b1, Cbuf, V_DIM);
    ln_relu_f16<<<B_DIM, blk, 0, stream>>>(Cbuf, q_g1, q_be1, Hbuf);
    gemm_wmma_f16<<<ggrid, blk, 0, stream>>>(Hbuf, WQ2T, q_b2, Cbuf, H_DIM);
    ln_relu_f16<<<B_DIM, blk, 0, stream>>>(Cbuf, q_g2, q_be2, Hbuf);
    gemm_wmma_f16<<<ggrid, blk, 0, stream>>>(Hbuf, WQ3T, q_b3, Cbuf, H_DIM);
    ln_relu_f16<<<B_DIM, blk, 0, stream>>>(Cbuf, q_g3, q_be3, Hbuf);

    q_head_sample<<<2, blk, 0, stream>>>(Hbuf, q_W, q_bias, q_marg, top_order, step,
                                         u, pa_mask, Vbuf, AF, LP);

    if (step < V_DIM - 1) {  // logFs[-1] is discarded by the reference
      gemm_wmma_f16<<<ggrid, blk, 0, stream>>>(AF, WF1T, f_b1, Cbuf, V_DIM);
      ln_relu_f16<<<B_DIM, blk, 0, stream>>>(Cbuf, f_g1, f_be1, Hbuf);
      gemm_wmma_f16<<<ggrid, blk, 0, stream>>>(Hbuf, WF2T, f_b2, Cbuf, H_DIM);
      ln_relu_f16<<<B_DIM, blk, 0, stream>>>(Cbuf, f_g2, f_be2, Hbuf);
      gemm_wmma_f16<<<ggrid, blk, 0, stream>>>(Hbuf, WF3T, f_b3, Cbuf, H_DIM);
      ln_relu_f16<<<B_DIM, blk, 0, stream>>>(Cbuf, f_g3, f_be3, Hbuf);
      f_head<<<2, blk, 0, stream>>>(Hbuf, f_w4, f_b4, LOGF + (size_t)step * B_DIM);
    }
  }

  // ---- terminal score + outputs ----
  float* outF = (float*)d_out;                 // [512*128] V, then [1] loss
  score_kernel<<<2, blk, 0, stream>>>(Vbuf, J, SCb, outF);
  loss_kernel<<<1, blk, 0, stream>>>(F0b, LOGF, SCb, LP, outF + B_DIM * V_DIM);
}
